// MultiHeadAttention_77180562309792
// MI455X (gfx1250) — matrix operand order repaired and checked by reading, not compiled, not run on hardware
//
#include <hip/hip_runtime.h>
#include <hip/hip_bf16.h>
#include <stdint.h>

#define DEVI __device__ __forceinline__

typedef __bf16 bf16_t;
typedef __attribute__((ext_vector_type(16))) __bf16 v16bf;
typedef __attribute__((ext_vector_type(8)))  float  v8f;
typedef __attribute__((ext_vector_type(8)))  unsigned int v8u;
typedef __attribute__((ext_vector_type(4)))  unsigned int v4u;

static constexpr int BATCH  = 2;
static constexpr int SEQ    = 2048;
static constexpr int DMODEL = 1024;
static constexpr int NH     = 16;
static constexpr int HDIM   = 64;
static constexpr int ROWS   = BATCH * SEQ; // 4096

// ---------- helpers ----------

DEVI unsigned short f32_to_bf16(float f) {
  unsigned int u = __builtin_bit_cast(unsigned int, f);
  unsigned int r = u + 0x7FFFu + ((u >> 16) & 1u); // round-to-nearest-even
  return (unsigned short)(r >> 16);
}

DEVI v16bf pack_ab(v4u lo, v4u hi) {
  v8u u;
  u[0] = lo[0]; u[1] = lo[1]; u[2] = lo[2]; u[3] = lo[3];
  u[4] = hi[0]; u[5] = hi[1]; u[6] = hi[2]; u[7] = hi[3];
  return __builtin_bit_cast(v16bf, u);
}

// A-operand tile (16x32 bf16, MxK) for v_wmma_*_16x16x32.
// base -> element (row0, k0) of a row-major [*, ld] bf16 matrix.
// ISA layout: lanes 0-15 row M=lane, VGPR0..3 = K (hi*8)+0..7, VGPR4..7 = K 16+(hi*8)+0..7.
DEVI v16bf load_a_tile(const bf16_t* base, int ld, int lane) {
  int r = lane & 15, hi = lane >> 4;
  const bf16_t* p = base + r * ld + hi * 8;
  v4u lo = *reinterpret_cast<const v4u*>(p);
  v4u hx = *reinterpret_cast<const v4u*>(p + 16);
  return pack_ab(lo, hx);
}

// B-operand tile (32x16 bf16, KxN) where B(k,n) = base[n*ld + k].
// ISA layout: lane holds column n=lane&15; VGPR0..3 = K (hi*8)+0..7, VGPR4..7 = K 16+(hi*8)+0..7.
DEVI v16bf load_b_tile(const bf16_t* base, int ld, int lane) {
  int n = lane & 15, hi = lane >> 4;
  const bf16_t* p = base + n * ld + hi * 8;
  v4u lo = *reinterpret_cast<const v4u*>(p);
  v4u hx = *reinterpret_cast<const v4u*>(p + 16);
  return pack_ab(lo, hx);
}

DEVI v8f wmma_bf16(v16bf a, v16bf b, v8f c) {
  // D = A*B + C ; emits v_wmma_f32_16x16x32_bf16
  return __builtin_amdgcn_wmma_f32_16x16x32_bf16(
      /*neg_a=*/false, a, /*neg_b=*/false, b,
      /*c_mod=*/(short)0, c, /*reuse_a=*/false, /*reuse_b=*/false);
}

// ---------- kernel 1: fp32 -> bf16 convert (x) and transposed-convert (weights) ----------

__global__ void cvt_x_kernel(const float* __restrict__ x, unsigned short* __restrict__ xb, int n4) {
  int i = blockIdx.x * blockDim.x + threadIdx.x;
  if (i >= n4) return;
  float4 v = reinterpret_cast<const float4*>(x)[i];
  uint2 o;
  o.x = (unsigned)f32_to_bf16(v.x) | ((unsigned)f32_to_bf16(v.y) << 16);
  o.y = (unsigned)f32_to_bf16(v.z) | ((unsigned)f32_to_bf16(v.w) << 16);
  reinterpret_cast<uint2*>(xb)[i] = o;
}

// wt[n*D + k] = bf16(w[k*D + n])  (K-contiguous per output column -> perfect B-operand loads)
__global__ void cvt_wT_kernel(const float* __restrict__ w, unsigned short* __restrict__ wt) {
  int i = blockIdx.x * blockDim.x + threadIdx.x; // i = k*D + n
  int k = i >> 10;
  int n = i & (DMODEL - 1);
  wt[n * DMODEL + k] = f32_to_bf16(w[i]);
}

// ---------- kernel 2: fused QKV projection ----------
// One wave computes a 32x64 tile of Q, K or V (2 A-tiles share 4 B-tiles -> 8 WMMA
// per k-step). Software-pipelined: next k-step operands are loaded before the
// current WMMA batch issues. Q,K stored [B,H,S,HD]; V stored transposed [B,H,HD,S].

__global__ __launch_bounds__(128) void qkv_proj_kernel(
    const bf16_t* __restrict__ xb,
    const bf16_t* __restrict__ wqT, const bf16_t* __restrict__ wkT, const bf16_t* __restrict__ wvT,
    unsigned short* __restrict__ Qh, unsigned short* __restrict__ Kh, unsigned short* __restrict__ Vt)
{
  int lane = threadIdx.x & 31;
  int unit = blockIdx.x * (blockDim.x >> 5) + (threadIdx.x >> 5); // 0..6143
  int rowstrip = unit & 127;           // 128 strips of 32 rows
  int colstrip = (unit >> 7) & 15;     // 16 strips of 64 cols
  int mat      = unit >> 11;           // 0=Q 1=K 2=V
  const bf16_t* wT = (mat == 0) ? wqT : (mat == 1) ? wkT : wvT;
  int row0 = rowstrip * 32, n0 = colstrip * 64;

  v8f c00 = {}, c01 = {}, c02 = {}, c03 = {};
  v8f c10 = {}, c11 = {}, c12 = {}, c13 = {};

  v16bf a0 = load_a_tile(xb + (size_t)(row0     ) * DMODEL, DMODEL, lane);
  v16bf a1 = load_a_tile(xb + (size_t)(row0 + 16) * DMODEL, DMODEL, lane);
  v16bf b0 = load_b_tile(wT + (size_t)(n0 +  0) * DMODEL, DMODEL, lane);
  v16bf b1 = load_b_tile(wT + (size_t)(n0 + 16) * DMODEL, DMODEL, lane);
  v16bf b2 = load_b_tile(wT + (size_t)(n0 + 32) * DMODEL, DMODEL, lane);
  v16bf b3 = load_b_tile(wT + (size_t)(n0 + 48) * DMODEL, DMODEL, lane);

  for (int k0 = 0; k0 < DMODEL; k0 += 32) {
    v16bf na0 = a0, na1 = a1, nb0 = b0, nb1 = b1, nb2 = b2, nb3 = b3;
    int kn = k0 + 32;
    if (kn < DMODEL) { // preload next k-step; independent of this step's WMMAs
      na0 = load_a_tile(xb + (size_t)(row0     ) * DMODEL + kn, DMODEL, lane);
      na1 = load_a_tile(xb + (size_t)(row0 + 16) * DMODEL + kn, DMODEL, lane);
      nb0 = load_b_tile(wT + (size_t)(n0 +  0) * DMODEL + kn, DMODEL, lane);
      nb1 = load_b_tile(wT + (size_t)(n0 + 16) * DMODEL + kn, DMODEL, lane);
      nb2 = load_b_tile(wT + (size_t)(n0 + 32) * DMODEL + kn, DMODEL, lane);
      nb3 = load_b_tile(wT + (size_t)(n0 + 48) * DMODEL + kn, DMODEL, lane);
    }
    c00 = wmma_bf16(a0, b0, c00);
    c01 = wmma_bf16(a0, b1, c01);
    c02 = wmma_bf16(a0, b2, c02);
    c03 = wmma_bf16(a0, b3, c03);
    c10 = wmma_bf16(a1, b0, c10);
    c11 = wmma_bf16(a1, b1, c11);
    c12 = wmma_bf16(a1, b2, c12);
    c13 = wmma_bf16(a1, b3, c13);
    a0 = na0; a1 = na1; b0 = nb0; b1 = nb1; b2 = nb2; b3 = nb3;
  }

  int hi = lane >> 4, nn = lane & 15;
  v8f cc[2][4] = {{c00, c01, c02, c03}, {c10, c11, c12, c13}};
#pragma unroll
  for (int rblk = 0; rblk < 2; ++rblk) {
#pragma unroll
    for (int t = 0; t < 4; ++t) {
#pragma unroll
      for (int v = 0; v < 8; ++v) {
        int row = row0 + rblk * 16 + v + 8 * hi;  // global token row in [0,4096)
        int col = n0 + t * 16 + nn;               // global model col in [0,1024)
        int b = row >> 11, s = row & (SEQ - 1);
        int h = col >> 6,  d = col & (HDIM - 1);
        unsigned short val = f32_to_bf16(cc[rblk][t][v]);
        if (mat == 0)      Qh[(((size_t)(b * NH + h) * SEQ) + s) * HDIM + d] = val;
        else if (mat == 1) Kh[(((size_t)(b * NH + h) * SEQ) + s) * HDIM + d] = val;
        else               Vt[(((size_t)(b * NH + h) * HDIM) + d) * SEQ + s] = val;
      }
    }
  }
}

// ---------- kernel 3: causal flash attention ----------
// One wave owns 16 query rows of one (b,h); streams 32 keys/iteration with
// online softmax. V-operand loads are issued BEFORE the softmax VALU block so
// the trans/VALU work hides their latency. K-tile loads are batched ahead of
// the 4 score WMMAs (single wait per group).

__global__ __launch_bounds__(128) void flash_attn_kernel(
    const bf16_t* __restrict__ Qh, const bf16_t* __restrict__ Kh, const bf16_t* __restrict__ Vt,
    unsigned short* __restrict__ ctx)
{
  __shared__ alignas(16) unsigned short pstage[4][16 * 32]; // per-wave P-tile transpose staging
  int lane = threadIdx.x & 31, wv = threadIdx.x >> 5;
  int unit = blockIdx.x * 4 + wv;     // 0..4095
  int qt = unit & 127;
  int h  = (unit >> 7) & 15;
  int b  = unit >> 11;
  int q0 = qt * 16;

  const bf16_t* q  = Qh + (size_t)(b * NH + h) * SEQ * HDIM;
  const bf16_t* kk = Kh + (size_t)(b * NH + h) * SEQ * HDIM;
  const bf16_t* vt = Vt + (size_t)(b * NH + h) * HDIM * SEQ;
  int hi = lane >> 4, nn = lane & 15;

  v16bf aq0 = load_a_tile(q + (size_t)q0 * HDIM,      HDIM, lane); // K = d 0..31
  v16bf aq1 = load_a_tile(q + (size_t)q0 * HDIM + 32, HDIM, lane); // K = d 32..63

  float m_i[8], l_i[8];
#pragma unroll
  for (int v = 0; v < 8; ++v) { m_i[v] = -__builtin_inff(); l_i[v] = 0.f; }
  v8f acc0 = {}, acc1 = {}, acc2 = {}, acc3 = {};

  const float csc = 0.125f * 1.44269504088896340736f; // 1/sqrt(64) * log2(e)
  int kend = q0 + 16;                                  // causal: keys < q0+16
  for (int kb = 0; kb < kend; kb += 32) {
    if (kb + 32 < kend) {
      __builtin_prefetch(kk + (size_t)(kb + 32) * HDIM, 0, 3); // global_prefetch_b8
      __builtin_prefetch(kk + (size_t)(kb + 48) * HDIM, 0, 3);
    }
    v8f zero = {};
    // batch all 4 K-tiles, then 4 score WMMAs
    v16bf bk00 = load_b_tile(kk + (size_t)(kb     ) * HDIM,      HDIM, lane);
    v16bf bk01 = load_b_tile(kk + (size_t)(kb     ) * HDIM + 32, HDIM, lane);
    v16bf bk10 = load_b_tile(kk + (size_t)(kb + 16) * HDIM,      HDIM, lane);
    v16bf bk11 = load_b_tile(kk + (size_t)(kb + 16) * HDIM + 32, HDIM, lane);
    v8f s0 = wmma_bf16(aq0, bk00, zero);
    s0     = wmma_bf16(aq1, bk01, s0);
    v8f s1 = wmma_bf16(aq0, bk10, zero);
    s1     = wmma_bf16(aq1, bk11, s1);

    // issue V-tile loads now; softmax VALU below hides their latency
    v16bf bv0 = load_b_tile(vt + (size_t)( 0) * SEQ + kb, SEQ, lane);
    v16bf bv1 = load_b_tile(vt + (size_t)(16) * SEQ + kb, SEQ, lane);
    v16bf bv2 = load_b_tile(vt + (size_t)(32) * SEQ + kb, SEQ, lane);
    v16bf bv3 = load_b_tile(vt + (size_t)(48) * SEQ + kb, SEQ, lane);

    float alpha[8];
#pragma unroll
    for (int v = 0; v < 8; ++v) {
      int row  = q0 + v + 8 * hi;
      int col0 = kb + nn, col1 = col0 + 16;
      float t0 = (col0 <= row) ? s0[v] * csc : -__builtin_inff();
      float t1 = (col1 <= row) ? s1[v] * csc : -__builtin_inff();
      float mx = fmaxf(t0, t1);
#pragma unroll
      for (int m = 1; m <= 8; m <<= 1) mx = fmaxf(mx, __shfl_xor(mx, m, 32));
      float mnew = fmaxf(m_i[v], mx);
      alpha[v] = __builtin_amdgcn_exp2f(m_i[v] - mnew);
      m_i[v] = mnew;
      t0 = __builtin_amdgcn_exp2f(t0 - mnew);
      t1 = __builtin_amdgcn_exp2f(t1 - mnew);
      s0[v] = t0; s1[v] = t1;
      float rs = t0 + t1;
#pragma unroll
      for (int m = 1; m <= 8; m <<= 1) rs += __shfl_xor(rs, m, 32);
      l_i[v] = l_i[v] * alpha[v] + rs;
    }
#pragma unroll
    for (int v = 0; v < 8; ++v) {
      acc0[v] *= alpha[v]; acc1[v] *= alpha[v];
      acc2[v] *= alpha[v]; acc3[v] *= alpha[v];
    }

    // P (C-layout) -> LDS -> A-layout (cross-lane transpose through LDS)
    unsigned short* pp = pstage[wv];
#pragma unroll
    for (int v = 0; v < 8; ++v) {
      int mrow = v + 8 * hi;
      pp[mrow * 32 + nn]      = f32_to_bf16(s0[v]);
      pp[mrow * 32 + 16 + nn] = f32_to_bf16(s1[v]);
    }
    __builtin_amdgcn_wave_barrier(); // wave-synchronous; DS ops in-order within wave
    v16bf pa = load_a_tile(reinterpret_cast<const bf16_t*>(pp), 32, lane);
    __builtin_amdgcn_wave_barrier();

    // acc += P(16x32) @ V(32x64); Vt[d][s] makes B-operand loads contiguous
    acc0 = wmma_bf16(pa, bv0, acc0);
    acc1 = wmma_bf16(pa, bv1, acc1);
    acc2 = wmma_bf16(pa, bv2, acc2);
    acc3 = wmma_bf16(pa, bv3, acc3);
  }

  // normalize and store ctx as bf16 in [B*S, D] row-major (head-concat layout)
#pragma unroll
  for (int v = 0; v < 8; ++v) {
    float inv = 1.0f / l_i[v];
    int row = q0 + v + 8 * hi;
    size_t base = (size_t)(b * SEQ + row) * DMODEL + h * HDIM + nn;
    ctx[base +  0] = f32_to_bf16(acc0[v] * inv);
    ctx[base + 16] = f32_to_bf16(acc1[v] * inv);
    ctx[base + 32] = f32_to_bf16(acc2[v] * inv);
    ctx[base + 48] = f32_to_bf16(acc3[v] * inv);
  }
}

// ---------- kernel 4: output projection + bias (fp32 out) ----------
// Same 32x64 register tile + software pipelining as the QKV GEMM.

__global__ __launch_bounds__(128) void out_proj_kernel(
    const bf16_t* __restrict__ cb, const bf16_t* __restrict__ woT,
    const float* __restrict__ bo, float* __restrict__ out)
{
  int lane = threadIdx.x & 31;
  int unit = blockIdx.x * 4 + (threadIdx.x >> 5); // 0..2047
  int rowstrip = unit & 127;
  int colstrip = unit >> 7;
  int row0 = rowstrip * 32, n0 = colstrip * 64;

  v8f c00 = {}, c01 = {}, c02 = {}, c03 = {};
  v8f c10 = {}, c11 = {}, c12 = {}, c13 = {};

  v16bf a0 = load_a_tile(cb + (size_t)(row0     ) * DMODEL, DMODEL, lane);
  v16bf a1 = load_a_tile(cb + (size_t)(row0 + 16) * DMODEL, DMODEL, lane);
  v16bf b0 = load_b_tile(woT + (size_t)(n0 +  0) * DMODEL, DMODEL, lane);
  v16bf b1 = load_b_tile(woT + (size_t)(n0 + 16) * DMODEL, DMODEL, lane);
  v16bf b2 = load_b_tile(woT + (size_t)(n0 + 32) * DMODEL, DMODEL, lane);
  v16bf b3 = load_b_tile(woT + (size_t)(n0 + 48) * DMODEL, DMODEL, lane);

  for (int k0 = 0; k0 < DMODEL; k0 += 32) {
    v16bf na0 = a0, na1 = a1, nb0 = b0, nb1 = b1, nb2 = b2, nb3 = b3;
    int kn = k0 + 32;
    if (kn < DMODEL) {
      na0 = load_a_tile(cb + (size_t)(row0     ) * DMODEL + kn, DMODEL, lane);
      na1 = load_a_tile(cb + (size_t)(row0 + 16) * DMODEL + kn, DMODEL, lane);
      nb0 = load_b_tile(woT + (size_t)(n0 +  0) * DMODEL + kn, DMODEL, lane);
      nb1 = load_b_tile(woT + (size_t)(n0 + 16) * DMODEL + kn, DMODEL, lane);
      nb2 = load_b_tile(woT + (size_t)(n0 + 32) * DMODEL + kn, DMODEL, lane);
      nb3 = load_b_tile(woT + (size_t)(n0 + 48) * DMODEL + kn, DMODEL, lane);
    }
    c00 = wmma_bf16(a0, b0, c00);
    c01 = wmma_bf16(a0, b1, c01);
    c02 = wmma_bf16(a0, b2, c02);
    c03 = wmma_bf16(a0, b3, c03);
    c10 = wmma_bf16(a1, b0, c10);
    c11 = wmma_bf16(a1, b1, c11);
    c12 = wmma_bf16(a1, b2, c12);
    c13 = wmma_bf16(a1, b3, c13);
    a0 = na0; a1 = na1; b0 = nb0; b1 = nb1; b2 = nb2; b3 = nb3;
  }

  int hi = lane >> 4, nn = lane & 15;
  v8f cc[2][4] = {{c00, c01, c02, c03}, {c10, c11, c12, c13}};
#pragma unroll
  for (int rblk = 0; rblk < 2; ++rblk) {
#pragma unroll
    for (int v = 0; v < 8; ++v) {
      int row = row0 + rblk * 16 + v + 8 * hi;
      float* o = out + (size_t)row * DMODEL + n0 + nn;
      o[ 0] = cc[rblk][0][v] + bo[n0 + nn +  0];
      o[16] = cc[rblk][1][v] + bo[n0 + nn + 16];
      o[32] = cc[rblk][2][v] + bo[n0 + nn + 32];
      o[48] = cc[rblk][3][v] + bo[n0 + nn + 48];
    }
  }
}

// ---------- launch ----------

extern "C" void kernel_launch(void* const* d_in, const int* in_sizes, int n_in,
                              void* d_out, int out_size, void* d_ws, size_t ws_size,
                              hipStream_t stream) {
  (void)in_sizes; (void)n_in; (void)out_size; (void)ws_size;
  const float* x  = (const float*)d_in[0];
  const float* Wq = (const float*)d_in[1];
  const float* Wk = (const float*)d_in[2];
  const float* Wv = (const float*)d_in[3];
  const float* Wo = (const float*)d_in[4];
  const float* bo = (const float*)d_in[5];
  float* out = (float*)d_out;

  char* ws = (char*)d_ws;
  size_t off = 0;
  auto alloc = [&](size_t bytes) {
    char* p = ws + off;
    off += (bytes + 255) & ~(size_t)255;
    return p;
  };
  unsigned short* xb  = (unsigned short*)alloc((size_t)ROWS * DMODEL * 2);   // 8 MB
  unsigned short* wqT = (unsigned short*)alloc((size_t)DMODEL * DMODEL * 2); // 2 MB
  unsigned short* wkT = (unsigned short*)alloc((size_t)DMODEL * DMODEL * 2);
  unsigned short* wvT = (unsigned short*)alloc((size_t)DMODEL * DMODEL * 2);
  unsigned short* woT = (unsigned short*)alloc((size_t)DMODEL * DMODEL * 2);
  unsigned short* Qh  = (unsigned short*)alloc((size_t)ROWS * DMODEL * 2);   // [B,H,S,HD]
  unsigned short* Kh  = (unsigned short*)alloc((size_t)ROWS * DMODEL * 2);   // [B,H,S,HD]
  unsigned short* Vt  = (unsigned short*)alloc((size_t)ROWS * DMODEL * 2);   // [B,H,HD,S]
  unsigned short* cb  = (unsigned short*)alloc((size_t)ROWS * DMODEL * 2);   // ctx [B*S,D]
  // total ~48 MB of workspace

  int n4 = ROWS * DMODEL / 4;
  cvt_x_kernel<<<n4 / 256, 256, 0, stream>>>(x, xb, n4);
  int nw = DMODEL * DMODEL / 256;
  cvt_wT_kernel<<<nw, 256, 0, stream>>>(Wq, wqT);
  cvt_wT_kernel<<<nw, 256, 0, stream>>>(Wk, wkT);
  cvt_wT_kernel<<<nw, 256, 0, stream>>>(Wv, wvT);
  cvt_wT_kernel<<<nw, 256, 0, stream>>>(Wo, woT);

  // 3 mats * 128 rowstrips(32) * 16 colstrips(64) = 6144 waves / 4 per block
  qkv_proj_kernel<<<1536, 128, 0, stream>>>(
      (const bf16_t*)xb, (const bf16_t*)wqT, (const bf16_t*)wkT, (const bf16_t*)wvT,
      Qh, Kh, Vt);

  // B*H*(S/16) = 4096 waves / 4 per block
  flash_attn_kernel<<<1024, 128, 0, stream>>>(
      (const bf16_t*)Qh, (const bf16_t*)Kh, (const bf16_t*)Vt, cb);

  // 128 rowstrips(32) * 16 colstrips(64) = 2048 waves / 4 per block
  out_proj_kernel<<<512, 128, 0, stream>>>(
      (const bf16_t*)cb, (const bf16_t*)woT, bo, out);
}
